// TensorProductConvLayer_2095944040529
// MI455X (gfx1250) — matrix operand order repaired and checked by reading, not compiled, not run on hardware
//
#include <hip/hip_runtime.h>

// ---------------------------------------------------------------------------
// MI455X (gfx1250) fused TensorProductConvLayer
//   K0: zero accumulators, transpose+convert W1/W2 to bf16 (n-major, ld=128)
//   K1: H = relu(edge_attr @ W1 + b1)        -> v_wmma_f32_16x16x32_bf16
//   K2: per-16-edge tile: w = dlrelu(H@W2+b2) fused with tensor product,
//       dual WMMA accumulator chains, LDS accumulation (ds_add_f32),
//       scatter (global_atomic_add_f32), W2 prefetch (global_prefetch_b8)
//   K3: column statistics over nodes
//   K4: residual + batch-norm-style normalize -> d_out (fp32)
// ---------------------------------------------------------------------------

typedef __attribute__((ext_vector_type(16))) __bf16 v16bf;
typedef __attribute__((ext_vector_type(8)))  float  v8f;

#define NE   32768
#define NN   4096
#define NS_  48
#define NV_  16
#define NC   96              // NS + 3*NV
#define NEF_ 128
#define WNUM_ 4096
#define INV_SQRT3_F 0.57735026918962576f
#define ALPHA_F     0.125f   // 1/sqrt(64)
#define EPS_F       1e-5f

__device__ __forceinline__ v8f wmma_bf16(v16bf a, v16bf b, v8f c) {
  return __builtin_amdgcn_wmma_f32_16x16x32_bf16(false, a, false, b, (short)0, c,
                                                 false, false);
}

// A-fragment (16x32 bf16): lane half "hi" -> 16 contiguous K k0..k0+15
__device__ __forceinline__ v16bf load_afrag_f32(const float* __restrict__ p) {
  v16bf a;
#pragma unroll
  for (int i = 0; i < 8; ++i) a[i] = (__bf16)p[i];
#pragma unroll
  for (int i = 0; i < 8; ++i) a[8 + i] = (__bf16)p[8 + i];
  return a;
}
__device__ __forceinline__ v16bf load_afrag_bf16(const __bf16* __restrict__ p) {
  v16bf a;
#pragma unroll
  for (int i = 0; i < 8; ++i) { a[i] = p[i]; a[8 + i] = p[8 + i]; }
  return a;
}
// B-fragment (32x16 bf16): lane holds 16 contiguous K values
__device__ __forceinline__ v16bf load_bfrag_bf16(const __bf16* __restrict__ p) {
  v16bf b;
#pragma unroll
  for (int i = 0; i < 16; ++i) b[i] = p[i];
  return b;
}

// ---------------------------------------------------------------------------
__global__ __launch_bounds__(256) void init_kernel(
    const float* __restrict__ w1, const float* __restrict__ w2,
    __bf16* __restrict__ w1t, __bf16* __restrict__ w2t,
    float* __restrict__ node_acc, float* __restrict__ stats) {
  const int stride = gridDim.x * blockDim.x;
  const int t0 = blockIdx.x * blockDim.x + threadIdx.x;
  for (int i = t0; i < NN * NC; i += stride) node_acc[i] = 0.f;
  for (int i = t0; i < 112; i += stride) stats[i] = 0.f;
  for (int i = t0; i < NEF_ * NEF_; i += stride) {
    int n = i >> 7, k = i & 127;
    w1t[i] = (__bf16)w1[k * NEF_ + n];            // W1T[n][k] = W1[k][n]
  }
  for (int i = t0; i < WNUM_ * NEF_; i += stride) {
    int n = i >> 7, k = i & 127;
    w2t[i] = (__bf16)w2[(size_t)k * WNUM_ + n];   // W2T[n][k] = W2[k][n]
  }
}

// ---------------------------------------------------------------------------
// K1: H[e][j] = relu(edge_attr@W1 + b1), bf16 row-major. 1 block = 16 edges.
__global__ __launch_bounds__(128) void fc1_kernel(
    const float* __restrict__ ea, const __bf16* __restrict__ w1t,
    const float* __restrict__ b1, __bf16* __restrict__ H) {
  const int e0   = blockIdx.x * 16;
  const int lane = threadIdx.x & 31;
  const int wave = threadIdx.x >> 5;
  const int hi   = lane >> 4;
  const int lm   = lane & 15;
  const int n0   = wave * 32;                 // 4 waves cover 128 columns

  v8f acc0 = {}, acc1 = {};
#pragma unroll
  for (int s = 0; s < 4; ++s) {
    const int ka = s * 32 + hi * 16;
    const int kb = s * 32 + hi * 16;
    v16bf a  = load_afrag_f32(ea + (size_t)(e0 + lm) * NEF_ + ka);
    v16bf bA = load_bfrag_bf16(w1t + (size_t)(n0 + lm) * NEF_ + kb);
    v16bf bB = load_bfrag_bf16(w1t + (size_t)(n0 + 16 + lm) * NEF_ + kb);
    acc0 = wmma_bf16(a, bA, acc0);
    acc1 = wmma_bf16(a, bB, acc1);
  }
  const int nA = n0 + lm, nB = n0 + 16 + lm;
  const float biasA = b1[nA], biasB = b1[nB];
#pragma unroll
  for (int r = 0; r < 8; ++r) {
    const int e = e0 + r + hi * 8;            // D layout: VGPR r -> row r (+8 for hi lanes)
    float xA = acc0[r] + biasA; xA = xA > 0.f ? xA : 0.f;
    float xB = acc1[r] + biasB; xB = xB > 0.f ? xB : 0.f;
    H[(size_t)e * NEF_ + nA] = (__bf16)xA;
    H[(size_t)e * NEF_ + nB] = (__bf16)xB;
  }
}

// ---------------------------------------------------------------------------
// Tensor-product accumulation of one 16-column tile of w (kept as a macro so
// the __shared__ address space stays visible -> ds_add_f32, not flat atomics).
#define ACCUM_TILE(C0, W)                                                      \
  do {                                                                         \
    const int c0_ = (C0);                                                      \
    if (c0_ < 2304) {               /* w_ss: i=c/48, k=c%48 */                 \
      const int i_ = c0_ / 48, k_ = (c0_ % 48) + lm;                           \
      _Pragma("unroll") for (int r = 0; r < 8; ++r) {                          \
        const int e_ = r + hi * 8;                                             \
        atomicAdd(&s_acc[e_][k_], sh_ass[e_][i_] * (W)[r]);                    \
      }                                                                        \
    } else if (c0_ < 3072) {        /* w_vs */                                 \
      const int cc_ = c0_ - 2304;                                              \
      const int i_ = cc_ / 48, k_ = (cc_ % 48) + lm;                           \
      _Pragma("unroll") for (int r = 0; r < 8; ++r) {                          \
        const int e_ = r + hi * 8;                                             \
        atomicAdd(&s_acc[e_][k_], sh_dot[e_][i_] * (W)[r]);                    \
      }                                                                        \
    } else if (c0_ < 3840) {        /* w_sv: i=c/16, k=lm */                   \
      const int i_ = (c0_ - 3072) / 16;                                        \
      _Pragma("unroll") for (int r = 0; r < 8; ++r) {                          \
        const int e_ = r + hi * 8;                                             \
        atomicAdd(&t_sv[e_][lm], sh_xs[e_][i_] * (W)[r]);                      \
      }                                                                        \
    } else {                        /* w_vv */                                 \
      const int i_ = (c0_ - 3840) / 16;                                        \
      _Pragma("unroll") for (int r = 0; r < 8; ++r) {                          \
        const int e_ = r + hi * 8;                                             \
        atomicAdd(&t_vv[e_][lm][0], sh_xv[e_][i_][0] * (W)[r]);                \
        atomicAdd(&t_vv[e_][lm][1], sh_xv[e_][i_][1] * (W)[r]);                \
        atomicAdd(&t_vv[e_][lm][2], sh_xv[e_][i_][2] * (W)[r]);                \
      }                                                                        \
    }                                                                          \
  } while (0)

// K2: fused  w = dlrelu(H@W2+b2)  +  tensor product  +  scatter.
// 1 block = 16 edges, 8 waves; each wave streams 32 of the 256 column tiles,
// two tiles per iteration (dual independent WMMA chains).
__global__ __launch_bounds__(256) void tp_kernel(
    const __bf16* __restrict__ H, const __bf16* __restrict__ w2t,
    const float* __restrict__ b2, const int* __restrict__ src,
    const int* __restrict__ dst, const float* __restrict__ node_attr,
    const float* __restrict__ edge_sh, float* __restrict__ node_acc) {
  __shared__ float sh_xs[16][NS_];
  __shared__ float sh_ass[16][NS_];
  __shared__ float sh_dot[16][NV_];
  __shared__ float sh_xv[16][NV_][3];
  __shared__ float sh_ss[16];
  __shared__ float sh_sv[16][3];
  __shared__ int   sh_dst[16];
  __shared__ float s_acc[16][NS_];
  __shared__ float t_sv[16][NV_];
  __shared__ float t_vv[16][NV_][3];

  const int e0  = blockIdx.x * 16;
  const int tid = threadIdx.x;

  if (tid < 16) {
    const int e = e0 + tid;
    sh_ss[tid]    = edge_sh[(size_t)e * 4 + 0];
    sh_sv[tid][0] = edge_sh[(size_t)e * 4 + 1];
    sh_sv[tid][1] = edge_sh[(size_t)e * 4 + 2];
    sh_sv[tid][2] = edge_sh[(size_t)e * 4 + 3];
    sh_dst[tid]   = dst[e];
  }
  __syncthreads();

  for (int idx = tid; idx < 16 * NS_; idx += blockDim.x) {
    const int e = idx / NS_, i = idx % NS_;
    const int s = src[e0 + e];
    const float x = node_attr[(size_t)s * NC + i];
    sh_xs[e][i]  = x;
    sh_ass[e][i] = x * sh_ss[e];
    s_acc[e][i]  = 0.f;
  }
  for (int idx = tid; idx < 16 * NV_; idx += blockDim.x) {
    const int e = idx / NV_, i = idx % NV_;
    const int s = src[e0 + e];
    const float v0 = node_attr[(size_t)s * NC + NS_ + i * 3 + 0];
    const float v1 = node_attr[(size_t)s * NC + NS_ + i * 3 + 1];
    const float v2 = node_attr[(size_t)s * NC + NS_ + i * 3 + 2];
    sh_xv[e][i][0] = v0; sh_xv[e][i][1] = v1; sh_xv[e][i][2] = v2;
    sh_dot[e][i] = (v0 * sh_sv[e][0] + v1 * sh_sv[e][1] + v2 * sh_sv[e][2]) * INV_SQRT3_F;
    t_sv[e][i] = 0.f;
    t_vv[e][i][0] = 0.f; t_vv[e][i][1] = 0.f; t_vv[e][i][2] = 0.f;
  }
  __syncthreads();

  const int lane = tid & 31;
  const int wave = tid >> 5;     // 0..7
  const int hi   = lane >> 4;
  const int lm   = lane & 15;

  // A fragments of H for this edge tile (reused for all 32 column tiles)
  v16bf afr[4];
#pragma unroll
  for (int s = 0; s < 4; ++s)
    afr[s] = load_afrag_bf16(H + (size_t)(e0 + lm) * NEF_ + s * 32 + hi * 16);

  // tiles for this wave: nt = wave + 8*j, j = 0..31; two per iteration
  for (int j = 0; j < 32; j += 2) {
    const int cA = (wave + 8 * j) * 16;        // column tile A
    const int cB = cA + 128;                   // column tile B (j+1)
    // prefetch next iteration's W2T lines (256 columns ahead)
    if (cA + 256 < WNUM_)
      __builtin_prefetch(w2t + (size_t)(cA + 256 + lm) * NEF_, 0, 1);

    v8f accA = {}, accB = {};
#pragma unroll
    for (int s = 0; s < 4; ++s) {
      const int kb = s * 32 + hi * 16;
      v16bf bA = load_bfrag_bf16(w2t + (size_t)(cA + lm) * NEF_ + kb);
      v16bf bB = load_bfrag_bf16(w2t + (size_t)(cB + lm) * NEF_ + kb);
      accA = wmma_bf16(afr[s], bA, accA);
      accB = wmma_bf16(afr[s], bB, accB);
    }
    const float biasA = b2[cA + lm];
    const float biasB = b2[cB + lm];
    float wA[8], wB[8];
#pragma unroll
    for (int r = 0; r < 8; ++r) {
      const float xA = accA[r] + biasA;
      const float xB = accB[r] + biasB;
      wA[r] = (fabsf(xA) <= 10.f) ? xA : 0.01f * xA;   // dlrelu
      wB[r] = (fabsf(xB) <= 10.f) ? xB : 0.01f * xB;
    }
    ACCUM_TILE(cA, wA);
    ACCUM_TILE(cB, wB);
  }
  __syncthreads();

  // scatter messages into per-node accumulator
  for (int idx = tid; idx < 16 * NS_; idx += blockDim.x) {
    const int e = idx / NS_, k = idx % NS_;
    atomicAdd(&node_acc[(size_t)sh_dst[e] * NC + k], ALPHA_F * s_acc[e][k]);
  }
  for (int idx = tid; idx < 16 * NV_ * 3; idx += blockDim.x) {
    const int e = idx / 48, rem = idx % 48;
    const int k = rem / 3, m = rem % 3;
    const float val =
        ALPHA_F * (t_sv[e][k] * sh_sv[e][m] + sh_ss[e] * t_vv[e][k][m]);
    atomicAdd(&node_acc[(size_t)sh_dst[e] * NC + NS_ + k * 3 + m], val);
  }
}

// ---------------------------------------------------------------------------
// K3: column sums / sum-squares over nodes.  stats[0:48]=sum_s, [48:96]=sumsq_s,
//     [96:112]=sumsq_v.  128 blocks x 64 threads, 32 rows per block.
__global__ __launch_bounds__(64) void stats_kernel(
    const float* __restrict__ node_acc, const float* __restrict__ node_attr,
    float* __restrict__ stats) {
  const int c  = threadIdx.x;
  const int r0 = blockIdx.x * 32;
  if (c < NS_) {
    float s = 0.f, sq = 0.f;
    for (int r = r0; r < r0 + 32; ++r) {
      const float x = node_acc[(size_t)r * NC + c] + node_attr[(size_t)r * NC + c];
      s += x; sq += x * x;
    }
    atomicAdd(&stats[c], s);
    atomicAdd(&stats[NS_ + c], sq);
  } else {
    const int k = c - NS_;
    float sq = 0.f;
    for (int r = r0; r < r0 + 32; ++r) {
      const float* p = node_acc + (size_t)r * NC + NS_ + k * 3;
      const float* q = node_attr + (size_t)r * NC + NS_ + k * 3;
#pragma unroll
      for (int m = 0; m < 3; ++m) { const float x = p[m] + q[m]; sq += x * x; }
    }
    atomicAdd(&stats[96 + k], sq);
  }
}

// ---------------------------------------------------------------------------
__global__ __launch_bounds__(96) void finalize_kernel(
    const float* __restrict__ node_acc, const float* __restrict__ node_attr,
    const float* __restrict__ stats, const float* __restrict__ bnw,
    const float* __restrict__ bnb, float* __restrict__ out) {
  const int n = blockIdx.x;
  const int c = threadIdx.x;
  const float x = node_acc[(size_t)n * NC + c] + node_attr[(size_t)n * NC + c];
  float o;
  if (c < NS_) {
    const float mean = stats[c] * (1.f / (float)NN);
    const float var  = stats[NS_ + c] * (1.f / (float)NN) - mean * mean;
    o = (x - mean) * rsqrtf(var + EPS_F) * bnw[c] + bnb[c];
  } else {
    const int k = (c - NS_) / 3;
    const float vn = stats[96 + k] * (1.f / ((float)NN * 3.f));
    o = x * rsqrtf(vn + EPS_F) * bnw[NS_ + k];
  }
  out[(size_t)n * NC + c] = o;
}

// ---------------------------------------------------------------------------
extern "C" void kernel_launch(void* const* d_in, const int* in_sizes, int n_in,
                              void* d_out, int out_size, void* d_ws, size_t ws_size,
                              hipStream_t stream) {
  const float* node_attr = (const float*)d_in[0];
  const int*   eidx      = (const int*)d_in[1];   // (2, E)
  const float* edge_attr = (const float*)d_in[2];
  const float* edge_sh   = (const float*)d_in[3];
  const float* fc_w1     = (const float*)d_in[4];
  const float* fc_b1     = (const float*)d_in[5];
  const float* fc_w2     = (const float*)d_in[6];
  const float* fc_b2     = (const float*)d_in[7];
  const float* bnw       = (const float*)d_in[8];
  const float* bnb       = (const float*)d_in[9];
  float* out = (float*)d_out;

  // workspace layout (~10.6 MB total)
  char* ws = (char*)d_ws;
  __bf16* H        = (__bf16*)(ws);                       // E*128 bf16    = 8 MB
  __bf16* W2T      = (__bf16*)(ws + 8388608);             // 4096*128 bf16 = 1 MB
  __bf16* W1T      = (__bf16*)(ws + 9437184);             // 128*128 bf16  = 32 KB
  float*  node_acc = (float*) (ws + 9469952);             // 4096*96 f32   = 1.5 MB
  float*  stats    = (float*) (ws + 11042816);            // 112 f32

  const int* src = eidx;
  const int* dst = eidx + NE;

  init_kernel<<<1024, 256, 0, stream>>>(fc_w1, fc_w2, W1T, W2T, node_acc, stats);
  fc1_kernel<<<NE / 16, 128, 0, stream>>>(edge_attr, W1T, fc_b1, H);
  tp_kernel<<<NE / 16, 256, 0, stream>>>(H, W2T, fc_b2, src, dst, node_attr,
                                         edge_sh, node_acc);
  stats_kernel<<<128, 64, 0, stream>>>(node_acc, node_attr, stats);
  finalize_kernel<<<NN, 96, 0, stream>>>(node_acc, node_attr, stats, bnw, bnb, out);
}
